// QuantizedLinear_88175678587513
// MI455X (gfx1250) — matrix operand order repaired and checked by reading, not compiled, not run on hardware
//
#include <hip/hip_runtime.h>

typedef __attribute__((ext_vector_type(8)))  __bf16 v8bf;
typedef __attribute__((ext_vector_type(16))) __bf16 v16bf;
typedef __attribute__((ext_vector_type(8)))  float  v8f;

#define GS 32   // MX group size

// ---------------------------------------------------------------------------
// FP4 E2M1 grid rounding: {0, ±0.5, 1, 1.5, 2, 3, 4, 6}, RNE like jnp.round.
// ---------------------------------------------------------------------------
__device__ __forceinline__ float fp4_round(float y) {
  float a = fabsf(y);
  float step = (a < 2.0f) ? 0.5f : ((a < 4.0f) ? 1.0f : 2.0f);
  float q = fminf(rintf(a * (2.0f / (step + step))) * step, 6.0f);
  return (y < 0.0f) ? -q : q;
}

// floor(log2(amax)) - 2, exactly (frexp gives m in [0.5,1): floor(log2)=ex-1)
__device__ __forceinline__ int group_exp(float amax) {
  int ex;
  (void)frexpf(amax, &ex);
  return ex - 3;
}

// ---------------------------------------------------------------------------
// Activation MXFP4 quant-dequant -> bf16 (exact: grid * 2^e is bf16-exact)
// one thread per 32-element group
// ---------------------------------------------------------------------------
__global__ void mxfp4_act_quant(const float* __restrict__ x,
                                __bf16* __restrict__ xq,
                                float* __restrict__ act_exps,
                                long long ngroups) {
  long long gid = (long long)blockIdx.x * blockDim.x + threadIdx.x;
  if (gid >= ngroups) return;
  const float* p = x + gid * GS;
  float v[GS];
  float amax = 0.0f;
#pragma unroll
  for (int i = 0; i < GS; i += 4) {
    float4 t = *(const float4*)(p + i);
    v[i] = t.x; v[i + 1] = t.y; v[i + 2] = t.z; v[i + 3] = t.w;
    amax = fmaxf(amax, fmaxf(fmaxf(fabsf(t.x), fabsf(t.y)),
                             fmaxf(fabsf(t.z), fabsf(t.w))));
  }
  int ei = (amax > 0.0f) ? group_exp(amax) : 0;
  float scale = ldexpf(1.0f, ei);
  float inv   = ldexpf(1.0f, -ei);   // exact pow-2 reciprocal
  __bf16 q[GS];
#pragma unroll
  for (int i = 0; i < GS; ++i)
    q[i] = (__bf16)(fp4_round(v[i] * inv) * scale);
#pragma unroll
  for (int i = 0; i < GS; i += 8)
    *(v8bf*)(xq + gid * GS + i) = *(const v8bf*)(q + i);
  act_exps[gid] = (float)ei;
}

// ---------------------------------------------------------------------------
// Weight MXFP4 quant-dequant with (1+eps_eff) scale -> bf16 (<=2^-9 rel err)
// also emits eps_eff and w_exps outputs. one thread per group.
// ---------------------------------------------------------------------------
__global__ void mxfp4_w_quant(const float* __restrict__ w,
                              const float* __restrict__ eps,
                              const float* __restrict__ delta,
                              __bf16* __restrict__ wq,
                              float* __restrict__ eps_eff_out,
                              float* __restrict__ w_exps_out,
                              long long ngroups) {
  long long gid = (long long)blockIdx.x * blockDim.x + threadIdx.x;
  if (gid >= ngroups) return;
  float ee = eps[gid] + delta[gid];
  eps_eff_out[gid] = ee;
  const float* p = w + gid * GS;   // (DOUT,DIN) row-major: group gid contiguous
  float v[GS];
  float amax = 0.0f;
#pragma unroll
  for (int i = 0; i < GS; i += 4) {
    float4 t = *(const float4*)(p + i);
    v[i] = t.x; v[i + 1] = t.y; v[i + 2] = t.z; v[i + 3] = t.w;
    amax = fmaxf(amax, fmaxf(fmaxf(fabsf(t.x), fabsf(t.y)),
                             fmaxf(fabsf(t.z), fabsf(t.w))));
  }
  int ei = (amax > 0.0f) ? group_exp(amax) : 0;
  w_exps_out[gid] = (float)ei;
  float scale = ldexpf(1.0f, ei) * (1.0f + ee);
  __bf16 q[GS];
#pragma unroll
  for (int i = 0; i < GS; ++i)
    q[i] = (__bf16)(fp4_round(v[i] / scale) * scale);
#pragma unroll
  for (int i = 0; i < GS; i += 8)
    *(v8bf*)(wq + gid * GS + i) = *(const v8bf*)(q + i);
}

// ---------------------------------------------------------------------------
// bf16 WMMA GEMM: out[M,N] = Xq[M,K] * Wq[N,K]^T + bias[N]
// 256x128 block tile, BK=32, 8 waves in 4x2: each wave 64x64 (16 WMMA tiles).
// Tiles staged with global_load_async_to_lds_b128 (ASYNCcnt), double-buffered,
// one s_wait_asynccnt + one barrier per k-iteration.
// LDS stride 40 elements (20 dwords) -> conflict-free fragment b128 loads.
// ---------------------------------------------------------------------------
#define TM  256
#define TN  128
#define BK  32
#define LDT 40

__global__ __launch_bounds__(256) void wmma_gemm_bf16(
    const __bf16* __restrict__ A,    // M x K
    const __bf16* __restrict__ Bw,   // N x K
    const float* __restrict__ bias,
    float* __restrict__ out,
    int M, int N, int K) {
  __shared__ __bf16 lsA[2][TM * LDT];   // 2 x 20480 B
  __shared__ __bf16 lsB[2][TN * LDT];   // 2 x 10240 B

  const int tid   = threadIdx.x;
  const int lane  = tid & 31;
  const int wid   = tid >> 5;     // 0..7
  const int waveM = wid >> 1;     // 0..3  (64 rows each)
  const int waveN = wid & 1;      // 0..1  (64 cols each)

  const int bM = blockIdx.y * TM;
  const int bN = blockIdx.x * TN;

  const __bf16* gA = A  + (long long)bM * K;
  const __bf16* gB = Bw + (long long)bN * K;

  // per-thread async-copy coordinates: 16B chunks, 4 per row of 32 bf16
  const __bf16* pAg[4];
  unsigned offAl[4];
#pragma unroll
  for (int i = 0; i < 4; ++i) {
    int c = tid + i * 256;              // 1024 chunks for A (256 rows)
    int r = c >> 2;
    int col = (c & 3) * 8;
    pAg[i]  = gA + (long long)r * K + col;
    offAl[i] = (unsigned)(r * LDT + col) * 2u;
  }
  const __bf16* pBg[2];
  unsigned offBl[2];
#pragma unroll
  for (int i = 0; i < 2; ++i) {
    int c = tid + i * 256;              // 512 chunks for B (128 rows)
    int r = c >> 2;
    int col = (c & 3) * 8;
    pBg[i]  = gB + (long long)r * K + col;
    offBl[i] = (unsigned)(r * LDT + col) * 2u;
  }

  const unsigned ldsA0 = (unsigned)(size_t)&lsA[0][0];
  const unsigned ldsB0 = (unsigned)(size_t)&lsB[0][0];
  const unsigned ABUF  = TM * LDT * 2u;
  const unsigned BBUF  = TN * LDT * 2u;

  auto async_tile = [&](int kt, int buf) {
#pragma unroll
    for (int i = 0; i < 4; ++i) {
      unsigned l = ldsA0 + (unsigned)buf * ABUF + offAl[i];
      const __bf16* g = pAg[i] + kt * BK;
      asm volatile("global_load_async_to_lds_b128 %0, %1, off"
                   :: "v"(l), "v"(g) : "memory");
    }
#pragma unroll
    for (int i = 0; i < 2; ++i) {
      unsigned l = ldsB0 + (unsigned)buf * BBUF + offBl[i];
      const __bf16* g = pBg[i] + kt * BK;
      asm volatile("global_load_async_to_lds_b128 %0, %1, off"
                   :: "v"(l), "v"(g) : "memory");
    }
  };

  v8f acc[4][4];
#pragma unroll
  for (int i = 0; i < 4; ++i)
#pragma unroll
    for (int j = 0; j < 4; ++j)
      acc[i][j] = (v8f){0.f, 0.f, 0.f, 0.f, 0.f, 0.f, 0.f, 0.f};

  auto compute = [&](int buf) {
    const int abase = (lane < 16) ? 0 : 8;    // A: lo K chunk 0/8, hi at +16
    const int bbase = (lane < 16) ? 0 : 8;    // B: lo K chunk 0/8, hi at +16
    v16bf afrag[4];
#pragma unroll
    for (int mt = 0; mt < 4; ++mt) {
      const __bf16* pr = &lsA[buf][(waveM * 64 + mt * 16 + (lane & 15)) * LDT + abase];
      v8bf lo = *(const v8bf*)pr;
      v8bf hi = *(const v8bf*)(pr + 16);
      afrag[mt] = __builtin_shufflevector(lo, hi, 0, 1, 2, 3, 4, 5, 6, 7,
                                          8, 9, 10, 11, 12, 13, 14, 15);
    }
    v16bf bfrag[4];
#pragma unroll
    for (int nt = 0; nt < 4; ++nt) {
      const __bf16* pr = &lsB[buf][(waveN * 64 + nt * 16 + (lane & 15)) * LDT + bbase];
      v8bf lo = *(const v8bf*)pr;
      v8bf hi = *(const v8bf*)(pr + 16);
      bfrag[nt] = __builtin_shufflevector(lo, hi, 0, 1, 2, 3, 4, 5, 6, 7,
                                          8, 9, 10, 11, 12, 13, 14, 15);
    }
#pragma unroll
    for (int mt = 0; mt < 4; ++mt)
#pragma unroll
      for (int nt = 0; nt < 4; ++nt)
        acc[mt][nt] = __builtin_amdgcn_wmma_f32_16x16x32_bf16(
            false, afrag[mt], false, bfrag[nt], (short)0, acc[mt][nt],
            false, false);
  };

  const int KT = K / BK;
  async_tile(0, 0);
  asm volatile("s_wait_asynccnt 0x0" ::: "memory");
  __syncthreads();
  for (int kt = 0; kt < KT; ++kt) {
    int cur = kt & 1;
    if (kt + 1 < KT) async_tile(kt + 1, cur ^ 1);  // background copy
    compute(cur);
    if (kt + 1 < KT) {
      asm volatile("s_wait_asynccnt 0x0" ::: "memory");
      __syncthreads();   // next buffer filled, current buffer fully consumed
    }
  }

  // Epilogue: C layout — VGPR r: lanes 0-15 -> M=base+r, lanes 16-31 -> M=base+8+r
#pragma unroll
  for (int mt = 0; mt < 4; ++mt) {
#pragma unroll
    for (int nt = 0; nt < 4; ++nt) {
      int mrow0 = bM + waveM * 64 + mt * 16 + ((lane < 16) ? 0 : 8);
      int ncol  = bN + waveN * 64 + nt * 16 + (lane & 15);
      float bv = bias[ncol];
#pragma unroll
      for (int r = 0; r < 8; ++r)
        out[(long long)(mrow0 + r) * N + ncol] = acc[mt][nt][r] + bv;
    }
  }
}

// ---------------------------------------------------------------------------
extern "C" void kernel_launch(void* const* d_in, const int* in_sizes, int n_in,
                              void* d_out, int out_size, void* d_ws,
                              size_t ws_size, hipStream_t stream) {
  const float* x     = (const float*)d_in[0];
  const float* delta = (const float*)d_in[1];
  const float* w     = (const float*)d_in[2];
  const float* bias  = (const float*)d_in[3];
  const float* eps   = (const float*)d_in[4];

  const int DOUT = in_sizes[3];            // 4096
  const int DIN  = in_sizes[2] / DOUT;     // 4096
  const int M    = in_sizes[0] / DIN;      // 8192 = B*S
  const int G    = DIN / GS;               // 128

  float* out_lin  = (float*)d_out;                         // M*DOUT
  float* out_eps  = out_lin + (long long)M * DOUT;         // DOUT*G
  float* out_wexp = out_eps + (long long)DOUT * G;         // DOUT*G
  float* out_aexp = out_wexp + (long long)DOUT * G;        // M*G

  __bf16* xq = (__bf16*)d_ws;                              // M x DIN bf16
  __bf16* wq = xq + (long long)M * DIN;                    // DOUT x DIN bf16

  long long agroups = (long long)M * G;
  long long wgroups = (long long)DOUT * G;

  mxfp4_act_quant<<<(unsigned)((agroups + 255) / 256), 256, 0, stream>>>(
      x, xq, out_aexp, agroups);
  mxfp4_w_quant<<<(unsigned)((wgroups + 255) / 256), 256, 0, stream>>>(
      w, eps, delta, wq, out_eps, out_wexp, wgroups);

  dim3 grid(DOUT / TN, M / TM);
  wmma_gemm_bf16<<<grid, 256, 0, stream>>>(xq, wq, bias, out_lin, M, DOUT, DIN);
}
